// MoETransformerLayer_28406913695889
// MI455X (gfx1250) — matrix operand order repaired and checked by reading, not compiled, not run on hardware
//
#include <hip/hip_runtime.h>
#include <hip/hip_bf16.h>
#include <stddef.h>

// ---------------------------------------------------------------------------
// MoE transformer layer for MI455X (gfx1250), wave32, WMMA bf16 path.
// D=1024, F=4096, E=8, top-2, T=8192 tokens.
// Weights pre-packed into WMMA B-fragment tile layout (32B contiguous/lane).
// Expert FFN: 32-row M-tiles, K-chunked H in LDS (192KB), each B fragment
// feeds two WMMAs (two M-halves) to halve L2 weight traffic.
// ---------------------------------------------------------------------------

#define D_MODEL 1024
#define D_FF    4096
#define N_EXP   8
#define N_TOK   8192          // 4 * 2048
#define N_SLOT  (N_TOK * 2)   // top-2 => 16384 (token,expert) pairs
#define M_ROWS  32            // rows per FFN tile
#define F_CHUNK 2048          // K-chunk of H kept in LDS
#define MAX_TILES 528         // sum ceil(c_e/32) <= 512 + 8
#define SMEM_BYTES ((M_ROWS * D_MODEL + M_ROWS * F_CHUNK) * 2)  // 64KB+128KB

// packed tile: 32(K) x 16(N) bf16 = 512 elems; lane L owns elems [L*16, L*16+16)
#define TILE_ELEMS 512
#define W1_KT (D_MODEL / 32)   // 32
#define W1_NT (D_FF / 16)      // 256
#define W2_KT (D_FF / 32)      // 128
#define W2_NT (D_MODEL / 16)   // 64

typedef __attribute__((ext_vector_type(16))) __bf16 v16bf;
typedef __attribute__((ext_vector_type(8)))  __bf16 v8bf;
typedef __attribute__((ext_vector_type(4)))  __bf16 v4bf;
typedef __attribute__((ext_vector_type(8)))  float  v8f;

// ---- WMMA fragment loaders (contiguous k = 16*half + s in both A and B, wave32) ----------

// A (16x32 bf16) from row-major LDS, stride `lds` elements.
__device__ __forceinline__ v16bf load_frag_A(const __bf16* src, int lds, int k0, int lane) {
  int m    = lane & 15;
  int half = lane >> 4;
  const __bf16* p = src + (size_t)m * lds + k0 + half * 16;
  v8bf lo = *(const v8bf*)(p);
  v8bf hi = *(const v8bf*)(p + 8);
  return __builtin_shufflevector(lo, hi, 0, 1, 2, 3, 4, 5, 6, 7,
                                 8, 9, 10, 11, 12, 13, 14, 15);
}

// B from pre-packed tiles: one contiguous 32B load per lane.
__device__ __forceinline__ v16bf load_frag_B(const __bf16* tile_base, int kt) {
  return *(const v16bf*)(tile_base + (size_t)kt * TILE_ELEMS);
}

// gelu(x) = 0.5x(1+tanh(c(x+0.044715x^3))) == x * sigmoid(2c(x+0.044715x^3))
__device__ __forceinline__ float gelu_tanh(float v) {
  float u = 1.5957691216057308f * (v + 0.044715f * v * v * v);
  return v / (1.0f + __expf(-u));
}

// ---------------------------------------------------------------------------
// Kernel 0: repack expert weights f32 -> bf16 WMMA B-fragment tile layout.
// ---------------------------------------------------------------------------
__global__ __launch_bounds__(256) void prep_kernel(
    const float* __restrict__ W1, const float* __restrict__ W2,
    __bf16* __restrict__ W1p, __bf16* __restrict__ W2p,
    int* __restrict__ counts, float* __restrict__ prob_sum) {
  size_t gid = (size_t)blockIdx.x * blockDim.x + threadIdx.x;
  size_t s = gid * 4;                       // 4 consecutive packed elems
  const size_t total = (size_t)N_EXP * D_MODEL * D_FF;
  if (s < total) {
    {   // ---- W1 ----
      int j    = (int)(s & 15);
      int lane = (int)((s >> 4) & 31);
      int kt   = (int)((s >> 9) & 31);
      int nt   = (int)((s >> 14) & 255);
      int e    = (int)(s >> 22);
      int n    = nt * 16 + (lane & 15);
      int kb   = kt * 32 + ((lane >> 4) << 4) + j;
      v4bf o;
#pragma unroll
      for (int i = 0; i < 4; ++i)
        o[i] = (__bf16)W1[((size_t)e * D_MODEL + kb + i) * D_FF + n];
      *(v4bf*)(W1p + s) = o;
    }
    {   // ---- W2 ----
      int j    = (int)(s & 15);
      int lane = (int)((s >> 4) & 31);
      int kt   = (int)((s >> 9) & 127);
      int nt   = (int)((s >> 16) & 63);
      int e    = (int)(s >> 22);
      int n    = nt * 16 + (lane & 15);
      int kb   = kt * 32 + ((lane >> 4) << 4) + j;
      v4bf o;
#pragma unroll
      for (int i = 0; i < 4; ++i)
        o[i] = (__bf16)W2[((size_t)e * D_FF + kb + i) * D_MODEL + n];
      *(v4bf*)(W2p + s) = o;
    }
  }
  if (gid < N_EXP) { counts[gid] = 0; prob_sum[gid] = 0.0f; }
}

// ---------------------------------------------------------------------------
// Kernel 1: LayerNorm + router (softmax, top-2, renorm). One block per token.
// ---------------------------------------------------------------------------
__global__ __launch_bounds__(256) void ln_router_kernel(
    const float* __restrict__ x, const float* __restrict__ gamma,
    const float* __restrict__ beta, const float* __restrict__ Wr,
    const float* __restrict__ br,
    __bf16* __restrict__ Xn, int* __restrict__ tok_expert,
    float* __restrict__ tok_w, int* __restrict__ counts,
    float* __restrict__ prob_sum, float* __restrict__ out) {
  __shared__ float sN[D_MODEL];
  __shared__ float sRed[16];
  __shared__ float sL[N_EXP];
  __shared__ float sStat[2];

  const int t    = blockIdx.x;
  const int tid  = threadIdx.x;
  const int wave = tid >> 5;
  const int lane = tid & 31;

  const float* xr = x + (size_t)t * D_MODEL;
  float4 v = ((const float4*)xr)[tid];
  ((float4*)(out + (size_t)t * D_MODEL))[tid] = v;   // residual base

  float s  = v.x + v.y + v.z + v.w;
  float ss = v.x * v.x + v.y * v.y + v.z * v.z + v.w * v.w;
#pragma unroll
  for (int off = 16; off > 0; off >>= 1) {
    s  += __shfl_down(s, off, 32);
    ss += __shfl_down(ss, off, 32);
  }
  if (lane == 0) { sRed[wave] = s; sRed[8 + wave] = ss; }
  __syncthreads();
  if (tid == 0) {
    float S = 0.f, SS = 0.f;
    for (int i = 0; i < 8; ++i) { S += sRed[i]; SS += sRed[8 + i]; }
    float mu  = S * (1.0f / D_MODEL);
    float var = SS * (1.0f / D_MODEL) - mu * mu;
    sStat[0] = mu;
    sStat[1] = rsqrtf(var + 1e-5f);
  }
  __syncthreads();
  float mu = sStat[0], rs = sStat[1];
  float4 g = ((const float4*)gamma)[tid];
  float4 b = ((const float4*)beta)[tid];
  float n0 = (v.x - mu) * rs * g.x + b.x;
  float n1 = (v.y - mu) * rs * g.y + b.y;
  float n2 = (v.z - mu) * rs * g.z + b.z;
  float n3 = (v.w - mu) * rs * g.w + b.w;
  sN[tid * 4 + 0] = n0; sN[tid * 4 + 1] = n1;
  sN[tid * 4 + 2] = n2; sN[tid * 4 + 3] = n3;
  v4bf nb = { (__bf16)n0, (__bf16)n1, (__bf16)n2, (__bf16)n3 };
  *(v4bf*)(Xn + (size_t)t * D_MODEL + tid * 4) = nb;
  __syncthreads();

  float acc = 0.0f;
#pragma unroll 8
  for (int i = 0; i < 32; ++i) {
    int d = lane + i * 32;
    acc += sN[d] * Wr[d * N_EXP + wave];
  }
#pragma unroll
  for (int off = 16; off > 0; off >>= 1) acc += __shfl_down(acc, off, 32);
  if (lane == 0) sL[wave] = acc + br[wave];
  __syncthreads();

  if (tid == 0) {
    float m = sL[0];
    for (int e = 1; e < N_EXP; ++e) m = fmaxf(m, sL[e]);
    float p[N_EXP], sum = 0.f;
    for (int e = 0; e < N_EXP; ++e) { p[e] = __expf(sL[e] - m); sum += p[e]; }
    float inv = 1.0f / sum;
    int e0 = 0;
    for (int e = 1; e < N_EXP; ++e) if (p[e] > p[e0]) e0 = e;
    int e1 = (e0 == 0) ? 1 : 0;
    for (int e = 0; e < N_EXP; ++e) if (e != e0 && p[e] > p[e1]) e1 = e;
    float w0 = p[e0] / (p[e0] + p[e1]);
    float w1 = p[e1] / (p[e0] + p[e1]);
    tok_expert[2 * t] = e0; tok_expert[2 * t + 1] = e1;
    tok_w[2 * t] = w0;      tok_w[2 * t + 1] = w1;
    atomicAdd(&counts[e0], 1);
    atomicAdd(&counts[e1], 1);
    for (int e = 0; e < N_EXP; ++e) atomicAdd(&prob_sum[e], p[e] * inv);
  }
}

// ---------------------------------------------------------------------------
// Kernel 2: scan counts -> offsets, build 32-row M-tile table, aux loss.
// ---------------------------------------------------------------------------
__global__ void scan_kernel(const int* __restrict__ counts,
                            const float* __restrict__ prob_sum,
                            int* __restrict__ offsets, int* __restrict__ cursor,
                            int* __restrict__ tile_expert, int* __restrict__ tile_row,
                            int* __restrict__ tile_nval, int* __restrict__ n_tiles,
                            float* __restrict__ out_aux) {
  if (threadIdx.x == 0 && blockIdx.x == 0) {
    int off = 0, k = 0;
    for (int e = 0; e < N_EXP; ++e) {
      offsets[e] = off;
      cursor[e]  = 0;
      int c = counts[e];
      for (int i = 0; i < c; i += M_ROWS) {
        tile_expert[k] = e;
        tile_row[k]    = off + i;
        tile_nval[k]   = (c - i < M_ROWS) ? (c - i) : M_ROWS;
        ++k;
      }
      off += c;
    }
    offsets[N_EXP] = off;
    n_tiles[0] = k;
    float aux = 0.f;
    for (int e = 0; e < N_EXP; ++e) aux += (float)counts[e] * prob_sum[e];
    out_aux[0] = (float)N_EXP * aux / ((float)N_TOK * (float)N_TOK);
  }
}

// ---------------------------------------------------------------------------
// Kernel 3: scatter tokens into per-expert slot lists.
// ---------------------------------------------------------------------------
__global__ __launch_bounds__(256) void scatter_kernel(
    const int* __restrict__ tok_expert, const float* __restrict__ tok_w,
    const int* __restrict__ offsets, int* __restrict__ cursor,
    int* __restrict__ slot_token, float* __restrict__ slot_w) {
  int t = blockIdx.x * blockDim.x + threadIdx.x;
  if (t >= N_TOK) return;
#pragma unroll
  for (int k = 0; k < 2; ++k) {
    int e   = tok_expert[2 * t + k];
    int pos = atomicAdd(&cursor[e], 1);
    int s   = offsets[e] + pos;
    slot_token[s] = t;
    slot_w[s]     = tok_w[2 * t + k];
  }
}

// ---------------------------------------------------------------------------
// Kernel 4: fused expert FFN, 32-row tiles, K-chunked H.
// For p in {0,1}: phase1 computes H[:, p*2048 .. +2048) = gelu(A@W1+b1) in LDS;
// phase2 accumulates delta += H_chunk @ W2_chunk into 16 persistent v8f accs.
// Epilogue: out[token] += w * (delta + b2) via global f32 atomics.
// Each B fragment (32B/lane) feeds two WMMAs (M-halves) -> halved L2 traffic.
// ---------------------------------------------------------------------------
__global__ __launch_bounds__(256) void moe_ffn_kernel(
    const __bf16* __restrict__ Xn,
    const __bf16* __restrict__ W1p, const __bf16* __restrict__ W2p,
    const float* __restrict__ b1, const float* __restrict__ b2,
    const int* __restrict__ slot_token, const float* __restrict__ slot_w,
    const int* __restrict__ tile_expert, const int* __restrict__ tile_row,
    const int* __restrict__ tile_nval, const int* __restrict__ n_tiles,
    float* __restrict__ out) {
  extern __shared__ char smem[];
  __bf16* sA = (__bf16*)smem;                                  // 32 x 1024
  __bf16* sH = (__bf16*)(smem + (size_t)M_ROWS * D_MODEL * 2); // 32 x 2048
  __shared__ int   sTok[M_ROWS];
  __shared__ float sW[M_ROWS];

  const int bt = blockIdx.x;
  if (bt >= n_tiles[0]) return;
  const int e    = tile_expert[bt];
  const int row0 = tile_row[bt];
  const int nval = tile_nval[bt];

  const int tid  = threadIdx.x;
  const int wave = tid >> 5;
  const int lane = tid & 31;
  const int half = lane >> 4;

  if (tid < M_ROWS) {
    if (tid < nval) { sTok[tid] = slot_token[row0 + tid]; sW[tid] = slot_w[row0 + tid]; }
    else            { sTok[tid] = -1; sW[tid] = 0.0f; }
  }
  __syncthreads();

  // Gather A tile (32 x 1024 bf16) via async global->LDS b128 copies;
  // padding rows zero-filled with plain DS stores.
  for (int idx = tid; idx < M_ROWS * (D_MODEL / 8); idx += 256) {
    int r = idx >> 7;           // row
    int c = (idx & 127) * 8;    // col (8 bf16 = 16B)
    int tok = sTok[r];
    __bf16* dst = sA + (size_t)r * D_MODEL + c;
    if (tok >= 0) {
      const __bf16* src = Xn + (size_t)tok * D_MODEL + c;
      unsigned lds_off = (unsigned)(size_t)dst;   // LDS aperture: low 32 bits
      asm volatile("global_load_async_to_lds_b128 %0, %1, off"
                   :: "v"(lds_off), "v"((unsigned long long)(size_t)src)
                   : "memory");
    } else {
      *(uint4*)dst = make_uint4(0u, 0u, 0u, 0u);
    }
  }
  asm volatile("s_wait_asynccnt 0x0" ::: "memory");
  __syncthreads();

  v8f acc2[16];   // phase-2 accumulators: 8 n-subtiles x 2 M-halves
#pragma unroll
  for (int i = 0; i < 16; ++i) acc2[i] = (v8f){};

  for (int p = 0; p < 2; ++p) {
    // ---- Phase 1: H[:, p*F_CHUNK ..) = gelu(A @ W1 + b1) -> sH ------------
    for (int ns = 0; ns < 16; ++ns) {
      const int ncol = p * F_CHUNK + wave * 256 + ns * 16;   // column in D_FF
      const __bf16* pB = W1p +
          ((size_t)(e * W1_NT + (ncol >> 4)) * W1_KT) * TILE_ELEMS + lane * 16;
      v8f acc0 = (v8f){}, acc1 = (v8f){};
#pragma unroll 8
      for (int ks = 0; ks < W1_KT; ++ks) {
        if ((ks & 7) == 0)
          __builtin_prefetch(pB + (size_t)(ks + 8) * TILE_ELEMS, 0, 3);
        v16bf b  = load_frag_B(pB, ks);
        v16bf a0 = load_frag_A(sA, D_MODEL, ks * 32, lane);
        v16bf a1 = load_frag_A(sA + 16 * D_MODEL, D_MODEL, ks * 32, lane);
        acc0 = __builtin_amdgcn_wmma_f32_16x16x32_bf16(false, a0, false, b,
                                                       (short)0, acc0, false, false);
        acc1 = __builtin_amdgcn_wmma_f32_16x16x32_bf16(false, a1, false, b,
                                                       (short)0, acc1, false, false);
      }
      const int lc   = wave * 256 + ns * 16 + (lane & 15);   // col within chunk
      const float bias = b1[e * D_FF + ncol + (lane & 15)];
#pragma unroll
      for (int r = 0; r < 8; ++r) {
        sH[(size_t)(half * 8 + r) * F_CHUNK + lc]      = (__bf16)gelu_tanh(acc0[r] + bias);
        sH[(size_t)(16 + half * 8 + r) * F_CHUNK + lc] = (__bf16)gelu_tanh(acc1[r] + bias);
      }
    }
    __syncthreads();

    // ---- Phase 2 partial: delta += H_chunk @ W2[p*F_CHUNK.., :] ----------
#pragma unroll
    for (int ns = 0; ns < 8; ++ns) {
      const int ncol = wave * 128 + ns * 16;                 // column in D_MODEL
      const __bf16* pB = W2p +
          ((size_t)(e * W2_NT + (ncol >> 4)) * W2_KT + p * (F_CHUNK / 32)) *
              TILE_ELEMS + lane * 16;
      v8f a20 = acc2[ns * 2], a21 = acc2[ns * 2 + 1];
#pragma unroll 8
      for (int ks = 0; ks < F_CHUNK / 32; ++ks) {
        if ((ks & 7) == 0)
          __builtin_prefetch(pB + (size_t)(ks + 8) * TILE_ELEMS, 0, 3);
        v16bf b  = load_frag_B(pB, ks);
        v16bf a0 = load_frag_A(sH, F_CHUNK, ks * 32, lane);
        v16bf a1 = load_frag_A(sH + 16 * F_CHUNK, F_CHUNK, ks * 32, lane);
        a20 = __builtin_amdgcn_wmma_f32_16x16x32_bf16(false, a0, false, b,
                                                      (short)0, a20, false, false);
        a21 = __builtin_amdgcn_wmma_f32_16x16x32_bf16(false, a1, false, b,
                                                      (short)0, a21, false, false);
      }
      acc2[ns * 2] = a20; acc2[ns * 2 + 1] = a21;
    }
    __syncthreads();
  }

  // ---- Epilogue: out[token] += w * (delta + b2) ---------------------------
#pragma unroll
  for (int ns = 0; ns < 8; ++ns) {
    const int n = wave * 128 + ns * 16 + (lane & 15);
    const float bias = b2[e * D_MODEL + n];
#pragma unroll
    for (int h2 = 0; h2 < 2; ++h2) {
      v8f acc = acc2[ns * 2 + h2];
#pragma unroll
      for (int r = 0; r < 8; ++r) {
        int m   = h2 * 16 + half * 8 + r;
        int tok = sTok[m];
        if (tok >= 0) {
          float val = (acc[r] + bias) * sW[m];
          atomicAdd(&out[(size_t)tok * D_MODEL + n], val);
        }
      }
    }
  }
}

// ---------------------------------------------------------------------------
// Host launcher. Workspace layout (needs ~152 MB).
// ---------------------------------------------------------------------------
extern "C" void kernel_launch(void* const* d_in, const int* in_sizes, int n_in,
                              void* d_out, int out_size, void* d_ws, size_t ws_size,
                              hipStream_t stream) {
  const float* x     = (const float*)d_in[0];
  const float* gamma = (const float*)d_in[1];
  const float* beta  = (const float*)d_in[2];
  const float* Wr    = (const float*)d_in[3];
  const float* br    = (const float*)d_in[4];
  const float* W1    = (const float*)d_in[5];
  const float* b1    = (const float*)d_in[6];
  const float* W2    = (const float*)d_in[7];
  const float* b2    = (const float*)d_in[8];
  float* out = (float*)d_out;

  char* ws = (char*)d_ws;
  size_t off = 0;
  auto carve = [&](size_t bytes) {
    void* p = ws + off;
    off = (off + bytes + 255) & ~(size_t)255;
    return p;
  };
  const size_t wbytes = (size_t)N_EXP * D_MODEL * D_FF * 2;
  __bf16* W1p       = (__bf16*)carve(wbytes);
  __bf16* W2p       = (__bf16*)carve(wbytes);
  __bf16* Xn        = (__bf16*)carve((size_t)N_TOK * D_MODEL * 2);
  int*    tok_exp   = (int*)  carve(N_SLOT * sizeof(int));
  float*  tok_w     = (float*)carve(N_SLOT * sizeof(float));
  int*    slot_tok  = (int*)  carve(N_SLOT * sizeof(int));
  float*  slot_w    = (float*)carve(N_SLOT * sizeof(float));
  int*    counts    = (int*)  carve(N_EXP * sizeof(int));
  float*  prob_sum  = (float*)carve(N_EXP * sizeof(float));
  int*    cursor    = (int*)  carve(N_EXP * sizeof(int));
  int*    offsets   = (int*)  carve((N_EXP + 1) * sizeof(int));
  int*    tile_e    = (int*)  carve(MAX_TILES * sizeof(int));
  int*    tile_r    = (int*)  carve(MAX_TILES * sizeof(int));
  int*    tile_n    = (int*)  carve(MAX_TILES * sizeof(int));
  int*    n_tiles   = (int*)  carve(sizeof(int));
  (void)ws_size; (void)in_sizes; (void)n_in; (void)out_size;

  // 0) weight repack f32 -> bf16 WMMA tile layout (+ zero accumulators)
  {
    size_t elems = (size_t)N_EXP * D_MODEL * D_FF;   // per matrix
    int blocks = (int)(elems / 4 / 256);             // 32768, exact
    prep_kernel<<<blocks, 256, 0, stream>>>(W1, W2, W1p, W2p, counts, prob_sum);
  }
  // 1) layernorm + router + residual base
  ln_router_kernel<<<N_TOK, 256, 0, stream>>>(x, gamma, beta, Wr, br, Xn,
                                              tok_exp, tok_w, counts, prob_sum, out);
  // 2) scan + tile table + aux loss (out[T*D])
  scan_kernel<<<1, 32, 0, stream>>>(counts, prob_sum, offsets, cursor,
                                    tile_e, tile_r, tile_n, n_tiles,
                                    out + (size_t)N_TOK * D_MODEL);
  // 3) token scatter
  scatter_kernel<<<N_TOK / 256, 256, 0, stream>>>(tok_exp, tok_w, offsets, cursor,
                                                  slot_tok, slot_w);
  // 4) fused expert FFN (192KB dynamic LDS)
  hipFuncSetAttribute(reinterpret_cast<const void*>(moe_ffn_kernel),
                      hipFuncAttributeMaxDynamicSharedMemorySize, SMEM_BYTES);
  moe_ffn_kernel<<<MAX_TILES, 256, SMEM_BYTES, stream>>>(
      Xn, W1p, W2p, b1, b2, slot_tok, slot_w,
      tile_e, tile_r, tile_n, n_tiles, out);
}
